// LinearRNNLayer_13915694039799
// MI455X (gfx1250) — matrix operand order repaired and checked by reading, not compiled, not run on hardware
//
#include <hip/hip_runtime.h>
#include <cstdint>
#include <cstddef>

// Problem constants (match reference)
#define B_    8
#define T_    2048
#define D_    1024
#define M_    (B_ * T_)          // 16384 rows
#define DECAY 0.99f
#define EPS_  1e-5f

// GEMM tiling: block 128(M) x 256(N) x 32(K); 8 waves; wave tile 64x64
#define BM 128
#define BN 256
#define BK 32
#define LDSS 40                  // padded LDS stride (elements) = 80B rows, 16B aligned

typedef __bf16 bf16x16 __attribute__((ext_vector_type(16)));
typedef float  f32x8   __attribute__((ext_vector_type(8)));
typedef unsigned int u32x4 __attribute__((ext_vector_type(4)));
typedef int          i32x4 __attribute__((ext_vector_type(4)));
typedef int          i32x8 __attribute__((ext_vector_type(8)));

enum { EPI_GATE = 0, EPI_FUSE = 1, EPI_GELU = 2, EPI_RES = 3 };

#if __has_builtin(__builtin_amdgcn_tensor_load_to_lds) && __has_builtin(__builtin_amdgcn_s_wait_tensorcnt)
#define HAVE_TDM 1
#endif

__device__ __forceinline__ unsigned short f2bf(float f) {
    union { float f; unsigned int u; } v; v.f = f;
    unsigned int r = v.u + 0x7FFFu + ((v.u >> 16) & 1u);   // round-to-nearest-even
    return (unsigned short)(r >> 16);
}

// ---------------------------------------------------------------------------
// fp32 -> bf16 TRANSPOSING weight conversion: in (K x N) -> out (N x K)
// One-time cost; coalesced reads, strided writes.
// ---------------------------------------------------------------------------
__global__ __launch_bounds__(256)
void cvt_bf16_t_kernel(const float* __restrict__ in, unsigned short* __restrict__ out,
                       int K, int N) {
    int i = blockIdx.x * 256 + threadIdx.x;
    if (i < K * N) {
        int k = i / N, n = i - k * N;
        out[(size_t)n * K + k] = f2bf(in[i]);
    }
}

// ---------------------------------------------------------------------------
// LayerNorm over last dim (D_=1024); one block per row; optional f32/bf16 out
// ---------------------------------------------------------------------------
__global__ __launch_bounds__(256)
void ln_kernel(const float* __restrict__ x, const float* __restrict__ w,
               const float* __restrict__ b,
               float* __restrict__ out32, unsigned short* __restrict__ out16) {
    __shared__ float sh[512];
    const int row = blockIdx.x;
    const float* xr = x + (size_t)row * D_;
    float s = 0.f, s2 = 0.f;
    for (int i = threadIdx.x; i < D_; i += 256) { float v = xr[i]; s += v; s2 += v * v; }
    sh[threadIdx.x] = s; sh[256 + threadIdx.x] = s2;
    __syncthreads();
    for (int st = 128; st > 0; st >>= 1) {
        if (threadIdx.x < st) {
            sh[threadIdx.x]       += sh[threadIdx.x + st];
            sh[256 + threadIdx.x] += sh[256 + threadIdx.x + st];
        }
        __syncthreads();
    }
    const float mean = sh[0] * (1.0f / D_);
    const float var  = sh[256] * (1.0f / D_) - mean * mean;
    const float rstd = rsqrtf(var + EPS_);
    for (int i = threadIdx.x; i < D_; i += 256) {
        float hv = (xr[i] - mean) * rstd * w[i] + b[i];
        if (out32) out32[(size_t)row * D_ + i] = hv;
        if (out16) out16[(size_t)row * D_ + i] = f2bf(hv);
    }
}

// ---------------------------------------------------------------------------
// Fixed-decay scan: h_t = 0.99*h_{t-1} + v_t (zero init), in place over v.
// One thread per (b,d) channel; lane-adjacent d -> coalesced.
// Writes bf16 copy into concat buffer (M x 2D): fwd cols [0,D), bwd [D,2D).
// ---------------------------------------------------------------------------
__global__ __launch_bounds__(256)
void scan_kernel(float* __restrict__ v, unsigned short* __restrict__ catb, int dir) {
    const int c = blockIdx.x * 256 + threadIdx.x;   // 0 .. B_*D_-1
    const int b = c >> 10;
    const int d = c & (D_ - 1);
    const size_t base = (size_t)b * T_ * D_ + d;
    const size_t cb   = (size_t)b * T_ * (2 * D_) + (dir ? (D_ + d) : d);
    float h = 0.f;
    if (dir == 0) {
        #pragma unroll 4
        for (int t = 0; t < T_; t++) {
            size_t o = base + (size_t)t * D_;
            h = fmaf(DECAY, h, v[o]);
            v[o] = h;
            catb[cb + (size_t)t * (2 * D_)] = f2bf(h);
        }
    } else {
        #pragma unroll 4
        for (int t = T_ - 1; t >= 0; t--) {
            size_t o = base + (size_t)t * D_;
            h = fmaf(DECAY, h, v[o]);
            v[o] = h;
            catb[cb + (size_t)t * (2 * D_)] = f2bf(h);
        }
    }
}

#ifdef HAVE_TDM
// ---------------------------------------------------------------------------
// Issue one TDM 2D tile load (tile_rows x BK bf16) from a row-major matrix
// (row stride = K elements) into LDS with an extra 16B pad per 64B row,
// reproducing the LDSS=40 layout.  pad_interval code 3 = every 16 DWORDs,
// pad_amount code 3 = 4 DWORDs.
// ---------------------------------------------------------------------------
__device__ __forceinline__ void tdm_stage(const unsigned short* gsrc, unsigned lds_byte,
                                          int tile_rows, int K) {
    unsigned long long ga = (unsigned long long)(size_t)gsrc;
    u32x4 g0;
    g0[0] = 1u;                                   // count=1 (valid user descriptor)
    g0[1] = lds_byte;                             // LDS byte address
    g0[2] = (unsigned)(ga & 0xffffffffu);         // global addr [31:0]
    g0[3] = (unsigned)(ga >> 32) | (2u << 30);    // global addr [56:32] | type=2
    const unsigned td0 = (unsigned)K;             // tensor_dim0 (elements)
    const unsigned td1 = (unsigned)M_;            // tensor_dim1 (generous row bound)
    i32x8 g1;
    g1[0] = (int)((1u << 16)                      // data_size = 2 bytes
                | (1u << 20)                      // pad_enable
                | (3u << 22)                      // pad_interval: 16 DWORDs
                | (3u << 25));                    // pad_amount:   4 DWORDs
    g1[1] = (int)((td0 & 0xffffu) << 16);                         // tensor_dim0[15:0]
    g1[2] = (int)((td0 >> 16) | ((td1 & 0xffffu) << 16));         // dim0 hi | dim1 lo
    g1[3] = (int)((td1 >> 16) | ((unsigned)BK << 16));            // dim1 hi | tile_dim0
    g1[4] = (int)(unsigned)tile_rows;                             // tile_dim1 (tile_dim2=0)
    g1[5] = (int)(unsigned)K;                                     // tensor_dim0_stride lo
    g1[6] = 0;                                                    // stride hi | dim1_stride lo
    g1[7] = 0;
    i32x4 z4 = {0, 0, 0, 0};
#if __clang_major__ >= 23
    i32x8 z8 = {0, 0, 0, 0, 0, 0, 0, 0};
    __builtin_amdgcn_tensor_load_to_lds(g0, g1, z4, z4, z8, 0);
#else
    __builtin_amdgcn_tensor_load_to_lds(g0, g1, z4, z4, 0);
#endif
}
#endif

// ---------------------------------------------------------------------------
// bf16 GEMM: C = A(MxK) * W(KxN) + bias with W supplied pre-transposed
// (WT = N x K row-major).  fp32 accumulate via v_wmma_f32_16x16x32_bf16.
// 8 waves; wave tile 64x64 = 16 accumulators; double-buffered LDS staged by
// the Tensor Data Mover (wave0: A tile, wave1: B tile) when available.
// ---------------------------------------------------------------------------
template <int EPI>
__global__ __launch_bounds__(256)
void gemm_bf16_kernel(const unsigned short* __restrict__ A,
                      const unsigned short* __restrict__ WT,
                      const float* __restrict__ bias,
                      const float* __restrict__ aux0,
                      const float* __restrict__ aux1,
                      const float* __restrict__ aux2,
                      float* __restrict__ outF,
                      unsigned short* __restrict__ outH,
                      int M, int N, int K) {
    __shared__ __align__(16) unsigned short As[2][BM * LDSS];   // 20 KB
    __shared__ __align__(16) unsigned short Bs[2][BN * LDSS];   // 40 KB

    const int tid  = threadIdx.x;
    const int lane = tid & 31;
    const int wave = tid >> 5;
    const int wm   = wave & 1;     // 2 waves along M (2 x 64 = 128)
    const int wn   = wave >> 1;    // 4 waves along N (4 x 64 = 256)
    const int hl   = lane >> 4;    // lane half (0/1)
    const int l16  = lane & 15;

    const int bm = blockIdx.y * BM;
    const int bn = blockIdx.x * BN;

    f32x8 acc[4][4];
    #pragma unroll
    for (int i = 0; i < 4; i++)
        #pragma unroll
        for (int j = 0; j < 4; j++)
            #pragma unroll
            for (int e = 0; e < 8; e++)
                acc[i][j][e] = 0.0f;

    const int nk = K / BK;

#ifdef HAVE_TDM
    const unsigned asb0 = (unsigned)(size_t)(void*)&As[0][0];
    const unsigned asb1 = (unsigned)(size_t)(void*)&As[1][0];
    const unsigned bsb0 = (unsigned)(size_t)(void*)&Bs[0][0];
    const unsigned bsb1 = (unsigned)(size_t)(void*)&Bs[1][0];
    if (wave == 0)      tdm_stage(A  + (size_t)bm * K, asb0, BM, K);
    else if (wave == 1) tdm_stage(WT + (size_t)bn * K, bsb0, BN, K);
    if (wave < 2) __builtin_amdgcn_s_wait_tensorcnt(0);
#else
    {   // manual staging of buffer 0, kb=0: contiguous 16B row chunks
        #pragma unroll
        for (int i = 0; i < 2; i++) {
            int chunk = tid + i * 256, r = chunk >> 2, c = chunk & 3;
            *(uint4*)&As[0][r * LDSS + c * 8] =
                *(const uint4*)(A + (size_t)(bm + r) * K + c * 8);
        }
        #pragma unroll
        for (int i = 0; i < 4; i++) {
            int chunk = tid + i * 256, r = chunk >> 2, c = chunk & 3;
            *(uint4*)&Bs[0][r * LDSS + c * 8] =
                *(const uint4*)(WT + (size_t)(bn + r) * K + c * 8);
        }
    }
#endif
    __syncthreads();

    for (int kb = 0; kb < nk; kb++) {
        const int cur = kb & 1;
        if (kb + 1 < nk) {
            const int kn = (kb + 1) * BK;
#ifdef HAVE_TDM
            if (wave == 0)      tdm_stage(A  + (size_t)bm * K + kn, cur ? asb0 : asb1, BM, K);
            else if (wave == 1) tdm_stage(WT + (size_t)bn * K + kn, cur ? bsb0 : bsb1, BN, K);
#else
            #pragma unroll
            for (int i = 0; i < 2; i++) {
                int chunk = tid + i * 256, r = chunk >> 2, c = chunk & 3;
                *(uint4*)&As[cur ^ 1][r * LDSS + c * 8] =
                    *(const uint4*)(A + (size_t)(bm + r) * K + kn + c * 8);
            }
            #pragma unroll
            for (int i = 0; i < 4; i++) {
                int chunk = tid + i * 256, r = chunk >> 2, c = chunk & 3;
                *(uint4*)&Bs[cur ^ 1][r * LDSS + c * 8] =
                    *(const uint4*)(WT + (size_t)(bn + r) * K + kn + c * 8);
            }
#endif
        }

        union Frag { unsigned int u[8]; bf16x16 v; };
        // B fragments: lane l16 = column n; halves split K (0..15 / 16..31),
        // contiguous 32B runs in Bs[n][k] -> 2x ds_load_b128 each.
        Frag bfr[4];
        #pragma unroll
        for (int nt = 0; nt < 4; nt++) {
            const unsigned short* bp = &Bs[cur][(wn * 64 + nt * 16 + l16) * LDSS + hl * 16];
            #pragma unroll
            for (int j = 0; j < 8; j++)
                bfr[nt].u[j] = *(const unsigned int*)(bp + 2 * j);
        }
        #pragma unroll
        for (int mt = 0; mt < 4; mt++) {
            // A fragment: lane l16 = row m; half 0: K 0..15, half 1: 16..31
            Frag afr;
            const unsigned short* ap = &As[cur][(wm * 64 + mt * 16 + l16) * LDSS];
            const int kb0 = hl * 16;
            #pragma unroll
            for (int j = 0; j < 4; j++) afr.u[j]     = *(const unsigned int*)(ap + kb0 + 2 * j);
            #pragma unroll
            for (int j = 0; j < 4; j++) afr.u[4 + j] = *(const unsigned int*)(ap + kb0 + 8 + 2 * j);
            #pragma unroll
            for (int nt = 0; nt < 4; nt++)
                acc[mt][nt] = __builtin_amdgcn_wmma_f32_16x16x32_bf16(
                    false, afr.v, false, bfr[nt].v, (short)0, acc[mt][nt], false, false);
        }

#ifdef HAVE_TDM
        if (wave < 2) __builtin_amdgcn_s_wait_tensorcnt(0);
#endif
        __syncthreads();
    }

    // Epilogue. C layout: lanes 0-15 -> (M=e, N=lane); lanes 16-31 -> (M=e+8, N=lane-16)
    #pragma unroll
    for (int mt = 0; mt < 4; mt++) {
        #pragma unroll
        for (int nt = 0; nt < 4; nt++) {
            const int col = bn + wn * 64 + nt * 16 + l16;
            const float bv = bias[col];
            #pragma unroll
            for (int e = 0; e < 8; e++) {
                const int row = bm + wm * 64 + mt * 16 + hl * 8 + e;
                const size_t idx = (size_t)row * N + col;
                const float val = acc[mt][nt][e] + bv;
                if (EPI == EPI_GATE) {
                    float s = 1.0f / (1.0f + expf(-val));       // v = sig(hW+b)*h*mask
                    outF[idx] = s * aux0[idx] * aux1[row];
                } else if (EPI == EPI_FUSE) {
                    float beta = 1.0f / (1.0f + expf(-val));    // x2 = res + b*hf + (1-b)*hb
                    outF[idx] = aux2[idx] + beta * aux0[idx] + (1.0f - beta) * aux1[idx];
                } else if (EPI == EPI_GELU) {
                    float g = 0.5f * val * (1.0f + erff(val * 0.70710678118654752f));
                    outH[idx] = f2bf(g);
                } else {
                    outF[idx] = val + aux0[idx];                // out = x2 + ffn
                }
            }
        }
    }
}

// ---------------------------------------------------------------------------
// Host orchestration (~304 MB workspace, 256B-aligned slabs, reuse):
//   h32 (64MB) -> x2 ; h16 (32MB) -> LN2 out ; vf/vb (64MB, scans in place) ;
//   catb (64MB, M x 2D bf16) -> FFN1 activations ; bf16 weights^T (16MB).
// ---------------------------------------------------------------------------
extern "C" void kernel_launch(void* const* d_in, const int* in_sizes, int n_in,
                              void* d_out, int out_size, void* d_ws, size_t ws_size,
                              hipStream_t stream) {
    const float* x    = (const float*)d_in[0];
    const float* mask = (const float*)d_in[1];
    const float* ln1w = (const float*)d_in[2];
    const float* ln1b = (const float*)d_in[3];
    const float* ln2w = (const float*)d_in[4];
    const float* ln2b = (const float*)d_in[5];
    const float* Wgf  = (const float*)d_in[6];
    const float* bgf  = (const float*)d_in[7];
    const float* Wgb  = (const float*)d_in[8];
    const float* bgb  = (const float*)d_in[9];
    const float* Wfu  = (const float*)d_in[10];
    const float* bfu  = (const float*)d_in[11];
    const float* Wf1  = (const float*)d_in[12];
    const float* bf1  = (const float*)d_in[13];
    const float* Wf2  = (const float*)d_in[14];
    const float* bf2  = (const float*)d_in[15];
    float* out = (float*)d_out;

    const size_t M = M_, D = D_;
    char* ws = (char*)d_ws;
    size_t off = 0;
    auto give = [&](size_t bytes) -> void* {
        void* p = ws + off;
        off += (bytes + 255) & ~(size_t)255;
        return p;
    };

    float*          h32  = (float*)give(M * D * 4);              // h (f32), later x2
    unsigned short* h16  = (unsigned short*)give(M * D * 2);     // h (bf16), later LN2 out
    float*          vf   = (float*)give(M * D * 4);              // v_fwd -> h_fwd
    float*          vb   = (float*)give(M * D * 4);              // v_bwd -> h_bwd
    unsigned short* catb = (unsigned short*)give(M * 2 * D * 2); // [h_fwd|h_bwd] bf16 -> FFN1 act
    unsigned short* wgfT = (unsigned short*)give(D * D * 2);     // all weights stored N x K bf16
    unsigned short* wgbT = (unsigned short*)give(D * D * 2);
    unsigned short* wfuT = (unsigned short*)give(2 * D * D * 2);
    unsigned short* wf1T = (unsigned short*)give(D * 2 * D * 2);
    unsigned short* wf2T = (unsigned short*)give(2 * D * D * 2);

    // 1) weights -> bf16, transposed to N x K
    cvt_bf16_t_kernel<<<(int)((D * D + 255) / 256), 256, 0, stream>>>(Wgf, wgfT, D_, D_);
    cvt_bf16_t_kernel<<<(int)((D * D + 255) / 256), 256, 0, stream>>>(Wgb, wgbT, D_, D_);
    cvt_bf16_t_kernel<<<(int)((2 * D * D + 255) / 256), 256, 0, stream>>>(Wfu, wfuT, 2 * D_, D_);
    cvt_bf16_t_kernel<<<(int)((2 * D * D + 255) / 256), 256, 0, stream>>>(Wf1, wf1T, D_, 2 * D_);
    cvt_bf16_t_kernel<<<(int)((2 * D * D + 255) / 256), 256, 0, stream>>>(Wf2, wf2T, 2 * D_, D_);

    // 2) LN1 -> h32 + h16
    ln_kernel<<<M_, 256, 0, stream>>>(x, ln1w, ln1b, h32, h16);

    // 3) gate GEMMs with sigmoid*h*mask epilogue
    gemm_bf16_kernel<EPI_GATE><<<dim3(D_ / BN, M_ / BM), 256, 0, stream>>>(
        h16, wgfT, bgf, h32, mask, nullptr, vf, nullptr, M_, D_, D_);
    gemm_bf16_kernel<EPI_GATE><<<dim3(D_ / BN, M_ / BM), 256, 0, stream>>>(
        h16, wgbT, bgb, h32, mask, nullptr, vb, nullptr, M_, D_, D_);

    // 4) decay scans (in place) + bf16 concat
    scan_kernel<<<(B_ * D_) / 256, 256, 0, stream>>>(vf, catb, 0);
    scan_kernel<<<(B_ * D_) / 256, 256, 0, stream>>>(vb, catb, 1);

    // 5) fusion GEMM (K = 2D) -> x2 in h32
    gemm_bf16_kernel<EPI_FUSE><<<dim3(D_ / BN, M_ / BM), 256, 0, stream>>>(
        catb, wfuT, bfu, vf, vb, x, h32, nullptr, M_, D_, 2 * D_);

    // 6) LN2 -> bf16 (reuse h16)
    ln_kernel<<<M_, 256, 0, stream>>>(h32, ln2w, ln2b, nullptr, h16);

    // 7) FFN1 (N = 2D) + exact GELU -> bf16 (reuse catb)
    gemm_bf16_kernel<EPI_GELU><<<dim3(2 * D_ / BN, M_ / BM), 256, 0, stream>>>(
        h16, wf1T, bf1, nullptr, nullptr, nullptr, nullptr, catb, M_, 2 * D_, D_);

    // 8) FFN2 (K = 2D) + residual -> out
    gemm_bf16_kernel<EPI_RES><<<dim3(D_ / BN, M_ / BM), 256, 0, stream>>>(
        catb, wf2T, bf2, h32, nullptr, nullptr, out, nullptr, M_, D_, 2 * D_);

    (void)in_sizes; (void)n_in; (void)out_size; (void)ws_size; (void)mask;
}
